// PositionWiseFeedforwardLayer_18313740550504
// MI455X (gfx1250) — matrix operand order repaired and checked by reading, not compiled, not run on hardware
//
#include <hip/hip_runtime.h>

// Problem constants (from reference): B=8, S=4096, H=512, P=2048, N=4
#define TOK   32768
#define HDIM  512
#define PDIM  2048
#define NDOM  4

// Tiling
#define MT       128     // tokens per block
#define KP       64      // P-chunk width
#define XSTRIDE  536     // padded LDS row stride (bf16 elems); 1072B = 67*16, conflict-spread
#define HSTRIDE  72      // padded h_lds row stride (bf16 elems); 144B = 9*16

typedef __attribute__((ext_vector_type(16))) __bf16 v16bf;
typedef __attribute__((ext_vector_type(8)))  float  v8f;

union FragU { v16bf v; uint4 q[2]; };

__device__ __forceinline__ unsigned short f2bf(float f) {
    unsigned u = __builtin_bit_cast(unsigned, f);
    unsigned r = (u + 0x7FFFu + ((u >> 16) & 1u)) >> 16;   // round-to-nearest-even
    return (unsigned short)r;
}

// A-matrix 16x32 bf16 fragment (ISA 7.12.2): lane holds row = lane%16;
// elements 0..7 = K k0..k0+7, elements 8..15 = K k0+16..k0+23, k0 = 8*(lane>=16).
__device__ __forceinline__ v16bf load_a_frag(const unsigned short* mat, int row0,
                                             int stride, int kbase) {
    int lane = threadIdx.x & 31;
    const unsigned short* p = mat + (row0 + (lane & 15)) * stride + kbase + ((lane >> 4) << 3);
    FragU f;
    f.q[0] = *(const uint4*)(p);
    f.q[1] = *(const uint4*)(p + 16);
    return f.v;
}

// B-matrix 32x16 bf16 fragment: lane holds col = lane%16; elements 0..7 = K k0..k0+7,
// elements 8..15 = K k0+16..k0+23, k0 = 8*(lane>=16). Matrix stored column-major (K contiguous, colstride between cols).
__device__ __forceinline__ v16bf load_b_frag(const unsigned short* mat, long col0,
                                             long colstride, long kbase) {
    int lane = threadIdx.x & 31;
    const unsigned short* p = mat + (col0 + (lane & 15)) * colstride + kbase + ((lane >> 4) << 3);
    FragU f;
    f.q[0] = *(const uint4*)(p);
    f.q[1] = *(const uint4*)(p + 16);
    return f.v;
}

// ---------------- prep kernels ----------------

// x (f32, [tok][H]) -> bf16 same layout, 8 elems/thread
extern "C" __global__ void conv_x_kernel(const float* __restrict__ x,
                                         unsigned short* __restrict__ xbf) {
    size_t i = (size_t)blockIdx.x * blockDim.x + threadIdx.x;  // 8-elem groups
    const float4* xf = (const float4*)x;
    float4 a = xf[2 * i], b = xf[2 * i + 1];
    uint4 o;
    o.x = (unsigned)f2bf(a.x) | ((unsigned)f2bf(a.y) << 16);
    o.y = (unsigned)f2bf(a.z) | ((unsigned)f2bf(a.w) << 16);
    o.z = (unsigned)f2bf(b.x) | ((unsigned)f2bf(b.y) << 16);
    o.w = (unsigned)f2bf(b.z) | ((unsigned)f2bf(b.w) << 16);
    ((uint4*)xbf)[i] = o;
}

// W1 (f32, [n][h][p]) -> bf16 w1t [n][p][h]  (h contiguous = K-major for GEMM1 B)
extern "C" __global__ void conv_w1_kernel(const float* __restrict__ W1,
                                          unsigned short* __restrict__ w1t) {
    int idx = blockIdx.x * blockDim.x + threadIdx.x;    // over N*P*H, h fastest
    int h = idx & (HDIM - 1);
    int p = (idx >> 9) & (PDIM - 1);
    int n = idx >> 20;
    w1t[idx] = f2bf(W1[((size_t)n * HDIM + h) * PDIM + p]);
}

// W2 (f32, [n][p][h]) -> bf16 w2t [n][h][p]  (p contiguous = K-major for GEMM2 B)
extern "C" __global__ void conv_w2_kernel(const float* __restrict__ W2,
                                          unsigned short* __restrict__ w2t) {
    int idx = blockIdx.x * blockDim.x + threadIdx.x;    // over N*H*P, p fastest
    int p = idx & (PDIM - 1);
    int h = (idx >> 11) & (HDIM - 1);
    int n = idx >> 20;
    w2t[idx] = f2bf(W2[((size_t)n * PDIM + p) * HDIM + h]);
}

// Gate: logits = x@Wr + br; d = 0.9*softmax + 0.025. One thread per token.
extern "C" __global__ void gate_kernel(const float* __restrict__ x,
                                       const float* __restrict__ Wr,
                                       const float* __restrict__ br,
                                       float* __restrict__ dg) {
    int t = blockIdx.x * blockDim.x + threadIdx.x;
    if (t >= TOK) return;
    float l0 = br[0], l1 = br[1], l2 = br[2], l3 = br[3];
    const float4* xr = (const float4*)(x + (size_t)t * HDIM);
    const float4* wr = (const float4*)Wr;   // row h = float4 of N=4 logit weights
    for (int i = 0; i < HDIM / 4; ++i) {
        float4 xv = xr[i];
        float4 w0 = wr[4 * i + 0], w1 = wr[4 * i + 1], w2 = wr[4 * i + 2], w3 = wr[4 * i + 3];
        l0 += xv.x * w0.x + xv.y * w1.x + xv.z * w2.x + xv.w * w3.x;
        l1 += xv.x * w0.y + xv.y * w1.y + xv.z * w2.y + xv.w * w3.y;
        l2 += xv.x * w0.z + xv.y * w1.z + xv.z * w2.z + xv.w * w3.z;
        l3 += xv.x * w0.w + xv.y * w1.w + xv.z * w2.w + xv.w * w3.w;
    }
    float m = fmaxf(fmaxf(l0, l1), fmaxf(l2, l3));
    float e0 = __expf(l0 - m), e1 = __expf(l1 - m), e2 = __expf(l2 - m), e3 = __expf(l3 - m);
    float inv = 1.0f / (e0 + e1 + e2 + e3);
    float4 dv;
    dv.x = 0.9f * e0 * inv + 0.025f;
    dv.y = 0.9f * e1 * inv + 0.025f;
    dv.z = 0.9f * e2 * inv + 0.025f;
    dv.w = 0.9f * e3 * inv + 0.025f;
    *(float4*)(dg + (size_t)t * NDOM) = dv;
}

// ---------------- fused MoE FFN (WMMA bf16) ----------------
// Block: 128 tokens, 256 threads (8 wave32).
// Gate fusion: out = sum_n (d_n * relu(x W1_n + b1_n)) W2_n + sum_n d_n b2_n.
// The per-row gate is folded into h before GEMM2, so ONE accumulator serves all
// domains (no per-domain y), each W2 fragment is read once per block, each W1 fragment by 2 waves.
// GEMM1: wave w owns h col tile tp1=w&3, row tiles tmb..tmb+3 (tmb=4*(w>>2)):
//        per k-step 1 B frag + 4 A frags -> 4 wmma.
// GEMM2: wave w owns output cols [w*64, w*64+64) (4 col tiles) x all 128 rows
//        (8 row tiles): per k-step 4 B frags + 8 A frags -> 32 wmma.
extern "C" __global__ __launch_bounds__(256)
void moe_ffn_kernel(const unsigned short* __restrict__ xbf,
                    const unsigned short* __restrict__ w1t,
                    const unsigned short* __restrict__ w2t,
                    const float* __restrict__ b1,
                    const float* __restrict__ b2,
                    const float* __restrict__ dgate,
                    float* __restrict__ out) {
    extern __shared__ unsigned short smem[];
    unsigned short* x_lds = smem;                    // MT x XSTRIDE bf16
    unsigned short* h_lds = smem + MT * XSTRIDE;     // MT x HSTRIDE bf16

    const int tid  = threadIdx.x;
    const int lane = tid & 31;
    const int w    = tid >> 5;
    const int tok0 = blockIdx.x * MT;
    const int hi8  = (lane >> 4) << 3;   // +8 rows for lanes 16..31 in C layout
    const int ln15 = lane & 15;

    const int tp1 = w & 3;               // GEMM1 col tile within 64-wide h chunk
    const int tmb = (w >> 2) * 4;        // GEMM1 first row tile (4 consecutive)

    // Stage x tile into LDS (128 rows x 512 bf16), 16B chunks.
    for (int i = tid; i < MT * (HDIM / 8); i += 256) {
        int row = i >> 6;       // 64 chunks per row
        int c8  = i & 63;
        uint4 v = *(const uint4*)(xbf + (size_t)(tok0 + row) * HDIM + c8 * 8);
        *(uint4*)(x_lds + row * XSTRIDE + c8 * 8) = v;
    }
    __syncthreads();

    v8f yacc[8][4];
    {
        v8f z = {};
        #pragma unroll
        for (int rt = 0; rt < 8; ++rt)
            #pragma unroll
            for (int tc = 0; tc < 4; ++tc) yacc[rt][tc] = z;
    }

    for (int n = 0; n < NDOM; ++n) {
        const unsigned short* w1n = w1t + (size_t)n * PDIM * HDIM;
        const unsigned short* w2n = w2t + (size_t)n * HDIM * PDIM;

        // Per-row gate values for the h rows this wave produces (reused all chunks).
        float dr1[4][8];
        #pragma unroll
        for (int j = 0; j < 4; ++j)
            #pragma unroll
            for (int v = 0; v < 8; ++v)
                dr1[j][v] = dgate[(size_t)(tok0 + (tmb + j) * 16 + v + hi8) * NDOM + n];

        for (int pc = 0; pc < PDIM / KP; ++pc) {
            const int pbase = pc * KP;
            if (pc + 1 < PDIM / KP) {   // hint next chunk's weight columns toward L2/L0
                __builtin_prefetch(w1n + (size_t)(pbase + KP + tp1 * 16) * HDIM, 0, 1);
                __builtin_prefetch(w2n + (size_t)(w * 64) * PDIM + pbase + KP, 0, 1);
            }
            __syncthreads();   // previous chunk's h_lds reads complete

            // GEMM1: h = d_n * relu(x @ W1[:, pbase:pbase+64] + b1)
            v8f hc[4];
            {
                v8f z = {};
                #pragma unroll
                for (int j = 0; j < 4; ++j) hc[j] = z;
            }
            #pragma unroll
            for (int ks = 0; ks < HDIM / 32; ++ks) {
                v16bf b = load_b_frag(w1n, pbase + tp1 * 16, HDIM, ks * 32);
                #pragma unroll
                for (int j = 0; j < 4; ++j) {
                    v16bf a = load_a_frag(x_lds, (tmb + j) * 16, XSTRIDE, ks * 32);
                    hc[j] = __builtin_amdgcn_wmma_f32_16x16x32_bf16(
                                false, a, false, b, (short)0, hc[j], false, false);
                }
            }
            {
                float bias = b1[n * PDIM + pbase + tp1 * 16 + ln15];
                #pragma unroll
                for (int j = 0; j < 4; ++j) {
                    int hrow0 = (tmb + j) * 16 + hi8;
                    int hcol  = tp1 * 16 + ln15;
                    #pragma unroll
                    for (int v = 0; v < 8; ++v) {
                        float val = hc[j][v] + bias;
                        val = val > 0.0f ? val : 0.0f;
                        val *= dr1[j][v];
                        h_lds[(hrow0 + v) * HSTRIDE + hcol] = f2bf(val);
                    }
                }
            }
            __syncthreads();   // h_lds ready

            // GEMM2: yacc += h @ W2[pbase:pbase+64, :]. Wave: 128 rows x 64 cols.
            #pragma unroll
            for (int ks = 0; ks < KP / 32; ++ks) {
                v16bf bf2[4];
                #pragma unroll
                for (int tc = 0; tc < 4; ++tc)
                    bf2[tc] = load_b_frag(w2n, w * 64 + tc * 16, PDIM, pbase + ks * 32);
                #pragma unroll
                for (int rt = 0; rt < 8; ++rt) {
                    v16bf a = load_a_frag(h_lds, rt * 16, HSTRIDE, ks * 32);
                    #pragma unroll
                    for (int tc = 0; tc < 4; ++tc)
                        yacc[rt][tc] = __builtin_amdgcn_wmma_f32_16x16x32_bf16(
                                           false, a, false, bf2[tc], (short)0,
                                           yacc[rt][tc], false, false);
                }
            }
        }
    }

    // Epilogue: out = yacc + sum_n d_n * b2_n   (rank-1 bias term)
    float b2s[4][4];   // [tc][n]
    #pragma unroll
    for (int tc = 0; tc < 4; ++tc)
        #pragma unroll
        for (int nn = 0; nn < 4; ++nn)
            b2s[tc][nn] = b2[nn * HDIM + w * 64 + tc * 16 + ln15];

    #pragma unroll
    for (int rt = 0; rt < 8; ++rt) {
        float4 d4[8];
        #pragma unroll
        for (int v = 0; v < 8; ++v)
            d4[v] = *(const float4*)(dgate + (size_t)(tok0 + rt * 16 + v + hi8) * NDOM);
        #pragma unroll
        for (int tc = 0; tc < 4; ++tc) {
            #pragma unroll
            for (int v = 0; v < 8; ++v) {
                float s = d4[v].x * b2s[tc][0] + d4[v].y * b2s[tc][1] +
                          d4[v].z * b2s[tc][2] + d4[v].w * b2s[tc][3];
                out[(size_t)(tok0 + rt * 16 + v + hi8) * HDIM + w * 64 + tc * 16 + ln15] =
                    yacc[rt][tc][v] + s;
            }
        }
    }
}

extern "C" void kernel_launch(void* const* d_in, const int* in_sizes, int n_in,
                              void* d_out, int out_size, void* d_ws, size_t ws_size,
                              hipStream_t stream) {
    const float* x  = (const float*)d_in[0];
    const float* Wr = (const float*)d_in[1];
    const float* br = (const float*)d_in[2];
    const float* W1 = (const float*)d_in[3];
    const float* b1 = (const float*)d_in[4];
    const float* W2 = (const float*)d_in[5];
    const float* b2 = (const float*)d_in[6];

    float* out = (float*)d_out;                      // x_out: TOK*HDIM floats
    float* dg  = out + (size_t)TOK * HDIM;           // d:     TOK*NDOM floats

    unsigned short* xbf = (unsigned short*)d_ws;                  // TOK*HDIM bf16
    unsigned short* w1t = xbf + (size_t)TOK * HDIM;               // N*P*H bf16
    unsigned short* w2t = w1t + (size_t)NDOM * PDIM * HDIM;       // N*H*P bf16

    conv_x_kernel <<<(TOK * HDIM / 8) / 256, 256, 0, stream>>>(x, xbf);
    conv_w1_kernel<<<(NDOM * PDIM * HDIM) / 256, 256, 0, stream>>>(W1, w1t);
    conv_w2_kernel<<<(NDOM * HDIM * PDIM) / 256, 256, 0, stream>>>(W2, w2t);
    gate_kernel   <<<TOK / 256, 256, 0, stream>>>(x, Wr, br, dg);

    size_t smem_bytes = (size_t)(MT * XSTRIDE + MT * HSTRIDE) * sizeof(unsigned short);
    moe_ffn_kernel<<<TOK / MT, 256, smem_bytes, stream>>>(xbf, w1t, w2t, b1, b2, dg, out);
}
